// GroupedQueryAttention_78898549227746
// MI455X (gfx1250) — matrix operand order repaired and checked by reading, not compiled, not run on hardware
//
#include <hip/hip_runtime.h>
#include <cstdint>
#include <cstddef>

// ---------------------------------------------------------------------------
// CDNA5 (gfx1250) wave32 WMMA implementation of GQA + ALiBi + sliding window.
// All matmuls: v_wmma_f32_16x16x32_bf16, fragments loaded straight from
// global/L2 (no LDS staging, no barriers in GEMM hot loops).
// ---------------------------------------------------------------------------

typedef __attribute__((ext_vector_type(16))) __bf16 v16bf;
typedef __attribute__((ext_vector_type(8)))  float  v8f;

struct U4x2 { uint4 a, b; };

__device__ __forceinline__ unsigned short f32_to_bf16(float f) {
  unsigned u = __builtin_bit_cast(unsigned, f);
  u += 0x7FFFu + ((u >> 16) & 1u);          // round-to-nearest-even
  return (unsigned short)(u >> 16);
}

__device__ __forceinline__ v16bf make_frag(uint4 lo, uint4 hi) {
  U4x2 t{lo, hi};
  return __builtin_bit_cast(v16bf, t);
}

__device__ __forceinline__ v8f wmma_bf16(v16bf a, v16bf b, v8f c) {
  return __builtin_amdgcn_wmma_f32_16x16x32_bf16(false, a, false, b, (short)0, c,
                                                 false, false);
}

#define SEQ   2048
#define DHEAD 128
#define NQH   16
#define NKVH  4
#define WIN   1024
#define LDST  40   // padded LDS row stride (elements): 80B, 16B aligned, bank-clean

// ---------------------------------------------------------------------------
// Prep 1: elementwise f32 -> bf16 (8 elements / thread, 16B stores)
// ---------------------------------------------------------------------------
__global__ __launch_bounds__(256)
void cvt_bf16(const float* __restrict__ src, unsigned short* __restrict__ dst) {
  size_t i = ((size_t)blockIdx.x * 256 + threadIdx.x) * 8;
  float4 a = *(const float4*)(src + i);
  float4 b = *(const float4*)(src + i + 4);
  alignas(16) unsigned short pk[8];
  pk[0] = f32_to_bf16(a.x); pk[1] = f32_to_bf16(a.y);
  pk[2] = f32_to_bf16(a.z); pk[3] = f32_to_bf16(a.w);
  pk[4] = f32_to_bf16(b.x); pk[5] = f32_to_bf16(b.y);
  pk[6] = f32_to_bf16(b.z); pk[7] = f32_to_bf16(b.w);
  *(uint4*)(dst + i) = *(const uint4*)pk;
}

// ---------------------------------------------------------------------------
// Prep 2: W f32 [K][N] -> bf16 transposed Wt[N][K] (32x32 LDS tiles)
// ---------------------------------------------------------------------------
__global__ __launch_bounds__(256)
void transpose_cvt(const float* __restrict__ W, unsigned short* __restrict__ Wt,
                   int K, int N) {
  __shared__ unsigned short t[32][33];
  const int tx = threadIdx.x & 31;
  const int tg = threadIdx.x >> 5;          // 8 groups of 4 rows
  const int n0 = blockIdx.x * 32;
  const int k0 = blockIdx.y * 32;
#pragma unroll
  for (int r = 0; r < 4; ++r) {
    int k = tg * 4 + r;
    t[k][tx] = f32_to_bf16(W[(size_t)(k0 + k) * N + n0 + tx]);
  }
  __syncthreads();
#pragma unroll
  for (int r = 0; r < 4; ++r) {
    int n = tg * 4 + r;
    Wt[(size_t)(n0 + n) * K + k0 + tx] = t[tx][n];
  }
}

// ---------------------------------------------------------------------------
// GEMM: C[M,N] = A[M,K] (bf16 row-major) x Bt[N,K] (bf16, pre-transposed).
// LDS-free: fragments loaded directly from global (L2-resident), letting the
// scheduler pipeline global_load_b128 clauses against WMMAs via loadcnt.
// 256 thr = 8 waves; block tile 128x128; wave tile 32x64 = 2x4 WMMA.
// Epilogues: 0 = bf16 head-scatter [B][NH][S][D]
//            1 = bf16 transposed   [B][NH][D][S]   (for V^T)
//            2 = f32 row-major     [M][N]          (final output)
// ---------------------------------------------------------------------------
template<int EPI, int NH>
__global__ __launch_bounds__(256)
void gemm_dg_wmma(const unsigned short* __restrict__ A,
                  const unsigned short* __restrict__ Bt,
                  void* __restrict__ Cout,
                  int M, int N, int K) {
  const int lane = threadIdx.x & 31;
  const int wid  = threadIdx.x >> 5;
  const int lrow = lane & 15;
  const int hi   = lane >> 4;
  const int wm   = wid & 3;                 // 4 wave rows (32 M each)
  const int wn   = wid >> 2;                // 2 wave cols (64 N each)
  const int m0   = blockIdx.y * 128 + wm * 32;
  const int n0   = blockIdx.x * 128 + wn * 64;

  // Per-lane base pointers (row/column fragments are contiguous in K)
  const unsigned short* arow0 = A  + (size_t)(m0 + lrow)      * K;
  const unsigned short* arow1 = A  + (size_t)(m0 + 16 + lrow) * K;
  const unsigned short* brow[4];
#pragma unroll
  for (int in = 0; in < 4; ++in)
    brow[in] = Bt + (size_t)(n0 + in * 16 + lrow) * K;

  v8f acc[2][4];
#pragma unroll
  for (int im = 0; im < 2; ++im)
#pragma unroll
    for (int in = 0; in < 4; ++in) acc[im][in] = v8f{};

#pragma unroll 2
  for (int k0 = 0; k0 < K; k0 += 32) {
    const int ka = k0 + hi * 16;            // A-frag chunk base (lane half)
    const int kb = k0 + hi * 16;            // B-frag chunk base
    v16bf a0 = make_frag(*(const uint4*)(arow0 + ka),
                         *(const uint4*)(arow0 + ka + 8));
    v16bf a1 = make_frag(*(const uint4*)(arow1 + ka),
                         *(const uint4*)(arow1 + ka + 8));
#pragma unroll
    for (int in = 0; in < 4; ++in) {
      v16bf b = make_frag(*(const uint4*)(brow[in] + kb),
                          *(const uint4*)(brow[in] + kb + 8));
      acc[0][in] = wmma_bf16(a0, b, acc[0][in]);
      acc[1][in] = wmma_bf16(a1, b, acc[1][in]);
    }
  }

  // ---- epilogue ----
#pragma unroll
  for (int im = 0; im < 2; ++im) {
#pragma unroll
    for (int in = 0; in < 4; ++in) {
      v8f c = acc[im][in];
      int n = n0 + in * 16 + lrow;
      int mbase = m0 + im * 16 + 8 * hi;    // 8 contiguous rows per lane
      if (EPI == 0) {                       // bf16 head-scatter [B][NH][S][D]
        int h = n >> 7, d = n & 127;
        unsigned short* o = (unsigned short*)Cout;
#pragma unroll
        for (int r = 0; r < 8; ++r) {
          int m = mbase + r;
          int bb = m >> 11, s = m & 2047;
          o[((size_t)(bb * NH + h) * SEQ + s) * DHEAD + d] = f32_to_bf16(c[r]);
        }
      } else if (EPI == 1) {                // bf16 transposed [B][NH][D][S]
        int h = n >> 7, d = n & 127;
        int bb = mbase >> 11, s = mbase & 2047;
        alignas(16) unsigned short pk[8];
#pragma unroll
        for (int r = 0; r < 8; ++r) pk[r] = f32_to_bf16(c[r]);
        unsigned short* o = (unsigned short*)Cout;
        *(uint4*)&o[((size_t)(bb * NH + h) * DHEAD + d) * SEQ + s] =
            *(const uint4*)pk;
      } else {                              // f32 row-major
        float* o = (float*)Cout;
#pragma unroll
        for (int r = 0; r < 8; ++r)
          o[(size_t)(mbase + r) * N + n] = c[r];
      }
    }
  }
}

// ---------------------------------------------------------------------------
// Flash attention over sliding window [i-1024, i]. (ALiBi term is identically
// zero inside the causal region; the +1.0 window bias is softmax-shift-
// invariant; -1e9 entries underflow to exp()==0 — all exact.)
// One wave per 16-query tile; 32 keys/iter: 8 score WMMAs + 8 PV WMMAs.
// ---------------------------------------------------------------------------
__global__ __launch_bounds__(128)
void attn_wmma(const unsigned short* __restrict__ Q,   // [B][16][S][D]
               const unsigned short* __restrict__ Km,  // [B][4][S][D]
               const unsigned short* __restrict__ Vt,  // [B][4][D][S]
               unsigned short* __restrict__ O) {       // [B][S][16*D]
  __shared__ unsigned short Plds[4 * 16 * LDST];

  const int lane = threadIdx.x & 31;
  const int wid  = threadIdx.x >> 5;
  const int lrow = lane & 15;
  const int hi   = lane >> 4;
  const int i0   = (blockIdx.x * 4 + wid) * 16;   // query-tile base row
  const int h    = blockIdx.y;
  const int b    = blockIdx.z;
  const int kvh  = h >> 2;                        // 16 q-heads / 4 kv-heads

  const unsigned short* Qh  = Q  + ((size_t)(b * NQH  + h)   * SEQ) * DHEAD;
  const unsigned short* Kh  = Km + ((size_t)(b * NKVH + kvh) * SEQ) * DHEAD;
  const unsigned short* Vth = Vt + ((size_t)(b * NKVH + kvh) * DHEAD) * SEQ;

  // Preload Q fragments (4 chunks of K=32 along D)
  v16bf qf[4];
  {
    const unsigned short* qr = Qh + (size_t)(i0 + lrow) * DHEAD;
#pragma unroll
    for (int kc = 0; kc < 4; ++kc) {
      const unsigned short* p = qr + kc * 32 + hi * 16;
      qf[kc] = make_frag(*(const uint4*)p, *(const uint4*)(p + 8));
    }
  }

  v8f acc[8];
#pragma unroll
  for (int nt = 0; nt < 8; ++nt) acc[nt] = v8f{};
  float m_run[8], l_run[8];
#pragma unroll
  for (int r = 0; r < 8; ++r) { m_run[r] = -__builtin_inff(); l_run[r] = 0.0f; }

  const float scale = 0.08838834764831845f;  // 1/sqrt(128)
  int jstart = i0 - WIN; if (jstart < 0) jstart = 0;
  jstart &= ~31;

  for (int j0 = jstart; j0 <= i0 + 15; j0 += 32) {
    // ---- scores: two 16-key subtiles ----
    v8f s0 = v8f{}, s1 = v8f{};
#pragma unroll
    for (int kc = 0; kc < 4; ++kc) {
      const unsigned short* kb0 =
          Kh + (size_t)(j0 + lrow) * DHEAD + kc * 32 + hi * 16;
      s0 = wmma_bf16(qf[kc], make_frag(*(const uint4*)kb0, *(const uint4*)(kb0 + 8)), s0);
      const unsigned short* kb1 =
          Kh + (size_t)(j0 + 16 + lrow) * DHEAD + kc * 32 + hi * 16;
      s1 = wmma_bf16(qf[kc], make_frag(*(const uint4*)kb1, *(const uint4*)(kb1 + 8)), s1);
    }

    // ---- mask + online softmax (row = i0 + 8*hi + r, col = j0 + lrow [+16]) ----
    const int c0 = j0 + lrow, c1 = c0 + 16;
    float p0[8], p1[8];
#pragma unroll
    for (int r = 0; r < 8; ++r) {
      int mrow = i0 + 8 * hi + r;
      float v0 = s0[r] * scale;
      float v1 = s1[r] * scale;
      if (!(c0 <= mrow && c0 >= mrow - WIN)) v0 = -1.0e30f;
      if (!(c1 <= mrow && c1 >= mrow - WIN)) v1 = -1.0e30f;
      float rm = fmaxf(v0, v1);
#pragma unroll
      for (int off = 1; off < 16; off <<= 1)
        rm = fmaxf(rm, __shfl_xor(rm, off, 32));
      float mn = fmaxf(m_run[r], rm);
      float sc = __expf(m_run[r] - mn);       // exp(-inf)=0 on first block
      m_run[r] = mn;
      p0[r] = __expf(v0 - mn);
      p1[r] = __expf(v1 - mn);
      float rs = p0[r] + p1[r];
#pragma unroll
      for (int off = 1; off < 16; off <<= 1)
        rs += __shfl_xor(rs, off, 32);
      l_run[r] = l_run[r] * sc + rs;
#pragma unroll
      for (int nt = 0; nt < 8; ++nt) acc[nt][r] *= sc;
    }

    // ---- relayout P (C-layout -> A-layout) through per-wave LDS ----
    {
      unsigned short* pl = &Plds[wid * 16 * LDST];
#pragma unroll
      for (int r = 0; r < 8; ++r) {
        int prow = 8 * hi + r;
        pl[prow * LDST + lrow]      = f32_to_bf16(p0[r]);
        pl[prow * LDST + 16 + lrow] = f32_to_bf16(p1[r]);
      }
    }
    asm volatile("s_wait_dscnt 0" ::: "memory");   // same-wave LDS RAW fence

    const unsigned short* pp = &Plds[wid * 16 * LDST + lrow * LDST + hi * 16];
    v16bf pa = make_frag(*(const uint4*)pp, *(const uint4*)(pp + 8));

    // ---- P x V : V^T rows are contiguous in the key dimension ----
#pragma unroll
    for (int nt = 0; nt < 8; ++nt) {
      const unsigned short* vb =
          Vth + (size_t)(nt * 16 + lrow) * SEQ + j0 + hi * 16;
      acc[nt] = wmma_bf16(pa, make_frag(*(const uint4*)vb, *(const uint4*)(vb + 8)),
                          acc[nt]);
    }
  }

  // ---- normalize + store O[B][S][16*D] (bf16) ----
#pragma unroll
  for (int r = 0; r < 8; ++r) {
    float inv = 1.0f / l_run[r];
    int s = i0 + 8 * hi + r;
#pragma unroll
    for (int nt = 0; nt < 8; ++nt) {
      int col = h * DHEAD + nt * 16 + lrow;
      O[((size_t)b * SEQ + s) * (NQH * DHEAD) + col] =
          f32_to_bf16(acc[nt][r] * inv);
    }
  }
}

// ---------------------------------------------------------------------------
// Host launcher
// ---------------------------------------------------------------------------
extern "C" void kernel_launch(void* const* d_in, const int* in_sizes, int n_in,
                              void* d_out, int out_size, void* d_ws, size_t ws_size,
                              hipStream_t stream) {
  (void)in_sizes; (void)n_in; (void)out_size; (void)ws_size;
  const float* X  = (const float*)d_in[0];   // [2,2048,2048]
  const float* Wq = (const float*)d_in[1];   // [2048,2048]
  const float* Wk = (const float*)d_in[2];   // [2048,512]
  const float* Wv = (const float*)d_in[3];   // [2048,512]
  const float* Wo = (const float*)d_in[4];   // [2048,2048]
  float* out = (float*)d_out;                // [2,2048,2048] f32

  // bf16 workspace (~76 MB): activations + pre-converted/transposed weights
  unsigned short* ws = (unsigned short*)d_ws;
  const size_t QN  = (size_t)2 * NQH  * SEQ * DHEAD;  // 8.39M
  const size_t KN  = (size_t)2 * NKVH * SEQ * DHEAD;  // 2.10M
  const size_t XN  = (size_t)2 * SEQ * 2048;          // 8.39M
  const size_t WQN = (size_t)2048 * 2048;             // 4.19M
  const size_t WKN = (size_t)2048 * 512;              // 1.05M
  unsigned short* Qw  = ws;
  unsigned short* Kw  = Qw  + QN;
  unsigned short* Vtw = Kw  + KN;
  unsigned short* Ow  = Vtw + KN;
  unsigned short* Xbf = Ow  + QN;
  unsigned short* Wqt = Xbf + XN;
  unsigned short* Wkt = Wqt + WQN;
  unsigned short* Wvt = Wkt + WKN;
  unsigned short* Wot = Wvt + WKN;

  const int M = 2 * SEQ;  // 4096

  // ---- prep: convert + transpose (memory-bound, ~4us total) ----
  cvt_bf16<<<dim3((unsigned)(XN / (256 * 8))), 256, 0, stream>>>(X, Xbf);
  transpose_cvt<<<dim3(2048 / 32, 2048 / 32), 256, 0, stream>>>(Wq, Wqt, 2048, 2048);
  transpose_cvt<<<dim3(512 / 32, 2048 / 32), 256, 0, stream>>>(Wk, Wkt, 2048, 512);
  transpose_cvt<<<dim3(512 / 32, 2048 / 32), 256, 0, stream>>>(Wv, Wvt, 2048, 512);
  transpose_cvt<<<dim3(2048 / 32, 2048 / 32), 256, 0, stream>>>(Wo, Wot, 2048, 2048);

  // ---- projections (LDS-free WMMA GEMMs) ----
  gemm_dg_wmma<0, NQH><<<dim3(2048 / 128, M / 128), 256, 0, stream>>>(
      Xbf, Wqt, Qw, M, 2048, 2048);
  gemm_dg_wmma<0, NKVH><<<dim3(512 / 128, M / 128), 256, 0, stream>>>(
      Xbf, Wkt, Kw, M, 512, 2048);
  gemm_dg_wmma<1, NKVH><<<dim3(512 / 128, M / 128), 256, 0, stream>>>(
      Xbf, Wvt, Vtw, M, 512, 2048);

  // ---- attention ----
  attn_wmma<<<dim3(SEQ / 16 / 4, NQH, 2), dim3(128), 0, stream>>>(Qw, Kw, Vtw, Ow);

  // ---- output projection ----
  gemm_dg_wmma<2, 1><<<dim3(2048 / 128, M / 128), 256, 0, stream>>>(
      Ow, Wot, out, M, 2048, 2048);
}
